// NSF_AR_48198122995922
// MI455X (gfx1250) — matrix operand order repaired and checked by reading, not compiled, not run on hardware
//
#include <hip/hip_runtime.h>
#include <hip/hip_bf16.h>
#include <math.h>

// NSF autoregressive forward for MI455X (gfx1250, wave32, WMMA).
// Compute-bound: 15x (32768x512x512) GEMMs ~ 258 GFLOP vs ~20MB of data.
// f16 WMMA (v_wmma_f32_16x16x32_f16) with f32 accumulation for the big GEMMs.
// W2 GEMM: double-buffered LDS tiles filled by GLOBAL_LOAD_ASYNC_TO_LDS_B128
// (ASYNCcnt) so the copy overlaps ds_load+WMMA compute with no VGPR staging.

#define NSF_DIM   16
#define NSF_K     8
#define NSF_HID   512
#define NSF_P     23      // 3*K-1
#define NSF_L     15      // DIM-1

#define LDS_STRIDE 40     // 32 halves + 8 pad: 80B rows -> conflict-free, 16B aligned

typedef __attribute__((ext_vector_type(16))) _Float16 v16h;
typedef __attribute__((ext_vector_type(8)))  float    v8f;

// Async 16B global -> LDS copy (per active lane). Tracked with ASYNCcnt.
__device__ __forceinline__ void async_cp16(unsigned ldsByte, const void* gaddr) {
  asm volatile("global_load_async_to_lds_b128 %0, %1, off"
               :: "v"(ldsByte), "v"(gaddr) : "memory");
}
__device__ __forceinline__ void wait_async0() {
  asm volatile("s_wait_asynccnt 0x0" ::: "memory");
}

// ---------------------------------------------------------------------------
// Weight conversion kernels (run every call; deterministic).
// W2T[l][n][k] = (f16) W2[l][k][n]              (15 x 512 x 512)
// W3T[l][n][k] = n<23 ? (f16) W3[l][k][n] : 0   (15 x 32 x 512, zero-padded)
// ---------------------------------------------------------------------------
__global__ __launch_bounds__(256) void conv_w2_kernel(
    const float* __restrict__ W2, _Float16* __restrict__ W2T) {
  int idx = blockIdx.x * 256 + threadIdx.x;
  if (idx >= NSF_L * NSF_HID * NSF_HID) return;
  int l   = idx / (NSF_HID * NSF_HID);
  int rem = idx % (NSF_HID * NSF_HID);
  int n   = rem / NSF_HID;
  int k   = rem % NSF_HID;
  W2T[idx] = (_Float16)W2[(size_t)l * NSF_HID * NSF_HID + (size_t)k * NSF_HID + n];
}

__global__ __launch_bounds__(256) void conv_w3_kernel(
    const float* __restrict__ W3, _Float16* __restrict__ W3T) {
  int idx = blockIdx.x * 256 + threadIdx.x;
  if (idx >= NSF_L * 32 * NSF_HID) return;
  int l   = idx / (32 * NSF_HID);
  int rem = idx % (32 * NSF_HID);
  int n   = rem / NSF_HID;
  int k   = rem % NSF_HID;
  float v = (n < NSF_P) ? W3[(size_t)l * NSF_HID * NSF_P + (size_t)k * NSF_P + n] : 0.f;
  W3T[idx] = (_Float16)v;
}

// ---------------------------------------------------------------------------
// fc1: h1 = tanh(x[:, :i] @ W1[li, :i] + b1[li])   (i <= 15 MACs per element)
// ---------------------------------------------------------------------------
__global__ __launch_bounds__(256) void fc1_kernel(
    const float* __restrict__ x, const float* __restrict__ W1,
    const float* __restrict__ b1, _Float16* __restrict__ H,
    int li, int i, int N) {
  int t = blockIdx.x * 256 + threadIdx.x;
  if (t >= N * NSF_HID) return;
  int row = t >> 9;
  int col = t & (NSF_HID - 1);
  float acc = b1[li * NSF_HID + col];
  const float* w  = W1 + (size_t)li * NSF_L * NSF_HID + col;
  const float* xr = x + (size_t)row * NSF_DIM;
  for (int j = 0; j < i; j++)
    acc += xr[j] * w[(size_t)j * NSF_HID];
  H[(size_t)row * NSF_HID + col] = (_Float16)tanhf(acc);
}

// ---------------------------------------------------------------------------
// gemm512_tanh: Out(N x 512, f16) = tanh(A(N x 512, f16) @ W2 + bias)
// B supplied transposed (Bt[n][k], f16, rows of 512).
//
// Block: 256 threads / 8 waves -> 128 rows x 64 cols; grid (N/128, 8).
// Per 32-wide k-step:
//   1) kick async global->LDS copies of the NEXT tile into the alternate
//      buffer (no VGPR staging; the TDM-class copy overlaps compute)
//   2) ds_load fragments (A + 4xB) from the current buffer, 4 WMMAs
//   3) s_wait_asynccnt 0 + barrier.
//
// CDNA5 fragment layouts (ISA 7.12.2):
//   A 16x32 f16 : lane<16 -> row (lane&15), halves = K{0..7, 16..23}
//                 lane>=16 -> same row,     halves = K{8..15, 24..31}
//   B 32x16 f16 : lane holds column (lane&15); A and B loaded k-contiguous: lanes<16 K0..15, lanes>=16 K16..31
//   C 16x16 f32 : lane col = lane&15; VGPR r -> row r (lane<16) / r+8 (lane>=16)
// ---------------------------------------------------------------------------
__global__ __launch_bounds__(256) void gemm512_tanh_kernel(
    const _Float16* __restrict__ A, const _Float16* __restrict__ Bt,
    const float* __restrict__ bias, _Float16* __restrict__ Out) {
  __shared__ _Float16 As[2][128 * LDS_STRIDE];   // 2 x 10240B
  __shared__ _Float16 Bs[2][64 * LDS_STRIDE];    // 2 x  5120B

  const int tid  = threadIdx.x;
  const int wave = tid >> 5;
  const int lane = tid & 31;
  const int hh   = lane >> 4;       // half-wave: 0 or 1
  const int l16  = lane & 15;
  const int rowBase = blockIdx.x * 128;
  const int colBase = blockIdx.y * 64;

  // Cooperative copy assignments (per thread, fixed across k-steps)
  const int arow  = tid >> 1;            // 0..127
  const int apart = (tid & 1) * 16;      // 0 or 16 halves
  const int bn    = tid >> 2;            // 0..63
  const int bpart = (tid & 3) * 8;       // 0,8,16,24 halves

  const _Float16* gA = A  + (size_t)(rowBase + arow) * NSF_HID + apart;
  const _Float16* gB = Bt + (size_t)(colBase + bn)   * NSF_HID + bpart;
  // LDS byte offsets of this thread's copy destinations (buffer 0)
  const unsigned sA0 = (unsigned)(uintptr_t)(&As[0][0] + arow * LDS_STRIDE + apart);
  const unsigned sB0 = (unsigned)(uintptr_t)(&Bs[0][0] + bn   * LDS_STRIDE + bpart);
  const unsigned aszB = 128 * LDS_STRIDE * 2;   // bytes per A buffer
  const unsigned bszB = 64 * LDS_STRIDE * 2;    // bytes per B buffer
  const int asz = 128 * LDS_STRIDE;             // halves per A buffer
  const int bsz = 64 * LDS_STRIDE;              // halves per B buffer

  // Preload k-step 0 into buffer 0 (async, then wait + barrier)
  async_cp16(sA0,      gA);
  async_cp16(sA0 + 16, gA + 8);
  async_cp16(sB0,      gB);
  wait_async0();
  __syncthreads();

  v8f acc[4] = {};
  const _Float16* fragA0 = &As[0][0] + (wave * 16 + l16) * LDS_STRIDE + hh * 16;
  const _Float16* fragB0 = &Bs[0][0] + l16 * LDS_STRIDE + hh * 16;

#pragma unroll
  for (int kk = 0; kk < 16; kk++) {
    const int cur = kk & 1;

    // (1) Async-copy the next k-step tile into the alternate buffer.
    if (kk < 15) {
      const _Float16* ga = gA + (kk + 1) * 32;
      const _Float16* gb = gB + (kk + 1) * 32;
      const unsigned da = sA0 + (unsigned)(1 - cur) * aszB;
      const unsigned db = sB0 + (unsigned)(1 - cur) * bszB;
      async_cp16(da,      ga);
      async_cp16(da + 16, ga + 8);
      async_cp16(db,      gb);
    }

    // (2) Fragment ds_loads then 4 back-to-back WMMAs.
    v16h a, b[4];
    const _Float16* fa = fragA0 + cur * asz;
    *((uint4*)&a)       = *(const uint4*)(fa);
    *(((uint4*)&a) + 1) = *(const uint4*)(fa + 8);
#pragma unroll
    for (int t = 0; t < 4; t++) {
      const _Float16* fb = fragB0 + cur * bsz + t * 16 * LDS_STRIDE;
      *((uint4*)&b[t])       = *(const uint4*)(fb);
      *(((uint4*)&b[t]) + 1) = *(const uint4*)(fb + 8);
    }
#pragma unroll
    for (int t = 0; t < 4; t++)
      acc[t] = __builtin_amdgcn_wmma_f32_16x16x32_f16(
          false, a, false, b[t], (short)0, acc[t], false, false);

    // (3) Ensure this wave's async copies landed, then block barrier.
    if (kk < 15) wait_async0();
    __syncthreads();
  }

  // Epilogue: bias + tanh + f16 store
  const int row0 = rowBase + wave * 16 + hh * 8;
#pragma unroll
  for (int t = 0; t < 4; t++) {
    const int col = colBase + t * 16 + l16;
    const float bv = bias[col];
#pragma unroll
    for (int r = 0; r < 8; r++) {
      float v = tanhf(acc[t][r] + bv);
      Out[(size_t)(row0 + r) * NSF_HID + col] = (_Float16)v;
    }
  }
}

// ---------------------------------------------------------------------------
// gemm_w3: params(N x 23, f32) = A(N x 512, f16) @ W3 + b3
// Bt zero-padded to 32 columns. Wave: 16 rows x 32 cols (2 WMMA tiles).
// Grid: N/128 blocks x 256 threads.
// ---------------------------------------------------------------------------
__global__ __launch_bounds__(256) void gemm_w3_kernel(
    const _Float16* __restrict__ A, const _Float16* __restrict__ Bt,
    const float* __restrict__ bias, float* __restrict__ P) {
  const int wave = threadIdx.x >> 5;
  const int lane = threadIdx.x & 31;
  const int hh   = lane >> 4;
  const int l16  = lane & 15;
  const int rowBase = (blockIdx.x * 8 + wave) << 4;

  v8f acc[2] = {};

  const _Float16* ap0 = A  + (size_t)(rowBase + l16) * NSF_HID + hh * 16;
  const _Float16* bp0 = Bt + (size_t)l16 * NSF_HID + hh * 16;

  for (int k0 = 0; k0 < NSF_HID; k0 += 32) {
    v16h a, b[2];
    *((uint4*)&a)       = *(const uint4*)(ap0 + k0);
    *(((uint4*)&a) + 1) = *(const uint4*)(ap0 + k0 + 8);
#pragma unroll
    for (int t = 0; t < 2; t++) {
      const _Float16* bp = bp0 + k0 + (size_t)t * 16 * NSF_HID;
      *((uint4*)&b[t])       = *(const uint4*)(bp);
      *(((uint4*)&b[t]) + 1) = *(const uint4*)(bp + 8);
    }
#pragma unroll
    for (int t = 0; t < 2; t++)
      acc[t] = __builtin_amdgcn_wmma_f32_16x16x32_f16(
          false, a, false, b[t], (short)0, acc[t], false, false);
  }

  const int row0 = rowBase + hh * 8;
#pragma unroll
  for (int t = 0; t < 2; t++) {
    const int col = t * 16 + l16;
    if (col < NSF_P) {
      const float bv = bias[col];
#pragma unroll
      for (int r = 0; r < 8; r++)
        P[(size_t)(row0 + r) * NSF_P + col] = acc[t][r] + bv;
    }
  }
}

// ---------------------------------------------------------------------------
// Rational-quadratic spline (faithful to reference incl. double softmax /
// double softplus). One thread per row.
// ---------------------------------------------------------------------------
__device__ __forceinline__ float nsf_softplus(float v) {
  return (v > 20.f) ? v : log1pf(expf(v));
}

// raw[8] -> cumulative knots c[9] in [-3,3] and bin sizes b[8]
__device__ __forceinline__ void nsf_bins(const float* raw, float* c, float* b) {
  float t[8];
  float m = raw[0];
#pragma unroll
  for (int j = 1; j < 8; j++) m = fmaxf(m, raw[j]);
  float s = 0.f;
#pragma unroll
  for (int j = 0; j < 8; j++) { t[j] = expf(raw[j] - m); s += t[j]; }
  float inv = 6.f / s;                      // Wn = 2B*softmax(raw), B=3
#pragma unroll
  for (int j = 0; j < 8; j++) t[j] *= inv;
  float m2 = t[0];
#pragma unroll
  for (int j = 1; j < 8; j++) m2 = fmaxf(m2, t[j]);
  float s2 = 0.f;
#pragma unroll
  for (int j = 0; j < 8; j++) { t[j] = expf(t[j] - m2); s2 += t[j]; }
  float inv2 = 1.f / s2;
  float csum = 0.f;
  c[0] = -3.f;
#pragma unroll
  for (int j = 0; j < 8; j++) {             // widths = 0.001 + 0.992*softmax(Wn)
    float wj = 0.001f + 0.992f * t[j] * inv2;
    csum += wj;
    c[j + 1] = 6.f * csum - 3.f;
  }
  c[8] = 3.f;
#pragma unroll
  for (int j = 0; j < 8; j++) b[j] = c[j + 1] - c[j];
}

__global__ __launch_bounds__(256) void spline_kernel(
    const float* __restrict__ x, const float* __restrict__ params,
    const float* __restrict__ initp, int useInit, int dim, int N,
    float* __restrict__ out) {
  int row = blockIdx.x * 256 + threadIdx.x;
  if (row >= N) return;

  float p[NSF_P];
  const float* src = useInit ? initp : (params + (size_t)row * NSF_P);
#pragma unroll
  for (int j = 0; j < NSF_P; j++) p[j] = src[j];

  float cw[9], wbin[8], ch[9], hbin[8];
  nsf_bins(p, cw, wbin);          // Wr = p[0..7]
  nsf_bins(p + 8, ch, hbin);      // Hr = p[8..15]

  float d[9];
  const float CST = logf(expf(0.999f) - 1.f);
  float dend = 0.001f + nsf_softplus(CST);           // == 1.0
  d[0] = dend; d[8] = dend;
#pragma unroll
  for (int j = 1; j < 8; j++)                        // Dr = p[16..22]
    d[j] = 0.001f + nsf_softplus(nsf_softplus(p[15 + j]));

  float xv = x[(size_t)row * NSF_DIM + dim];
  bool inside = (xv >= -3.f) && (xv <= 3.f);
  float xc = fminf(fmaxf(xv, -3.f), 3.f);

  int idx = -1;
#pragma unroll
  for (int j = 0; j < 9; j++) {
    float edge = (j == 8) ? (3.f + 1e-6f) : cw[j];
    idx += (xc >= edge) ? 1 : 0;
  }
  idx = idx < 0 ? 0 : (idx > 7 ? 7 : idx);

  float in_cw = cw[idx],  in_w = wbin[idx];
  float in_ch = ch[idx],  in_h = hbin[idx];
  float delta = in_h / in_w;
  float dd0 = d[idx], dd1 = d[idx + 1];

  float theta = (xc - in_cw) / in_w;
  float tt  = theta * (1.f - theta);
  float num = in_h * (delta * theta * theta + dd0 * tt);
  float den = delta + (dd0 + dd1 - 2.f * delta) * tt;
  float z   = in_ch + num / den;
  float omt = 1.f - theta;
  float dnum = delta * delta * (dd1 * theta * theta + 2.f * delta * tt + dd0 * omt * omt);
  float ld  = logf(dnum) - 2.f * logf(den);

  out[(size_t)row * NSF_DIM + dim] = inside ? z : xv;
  size_t lo = (size_t)N * NSF_DIM + row;
  float lf = inside ? ld : 0.f;
  if (dim == 0) out[lo] = lf;
  else          out[lo] += lf;
}

// ---------------------------------------------------------------------------
// Host launcher. Workspace layout (all 256B aligned):
//   W2T f16 [15][512][512]   7.9 MB
//   W3T f16 [15][32][512]    0.5 MB
//   h1  f16 [N][512]        32  MB
//   h2  f16 [N][512]        32  MB
//   prm f32 [N][23]          3  MB
// ---------------------------------------------------------------------------
extern "C" void kernel_launch(void* const* d_in, const int* in_sizes, int n_in,
                              void* d_out, int out_size, void* d_ws, size_t ws_size,
                              hipStream_t stream) {
  const float* x     = (const float*)d_in[0];
  const float* initp = (const float*)d_in[1];
  const float* W1    = (const float*)d_in[2];
  const float* b1    = (const float*)d_in[3];
  const float* W2    = (const float*)d_in[4];
  const float* b2    = (const float*)d_in[5];
  const float* W3    = (const float*)d_in[6];
  const float* b3    = (const float*)d_in[7];
  float* out = (float*)d_out;
  const int N = in_sizes[0] / NSF_DIM;    // 32768

  char* ws = (char*)d_ws;
  size_t off = 0;
  auto take = [&](size_t bytes) {
    char* p = ws + off;
    off = (off + bytes + 255) & ~(size_t)255;
    return p;
  };
  _Float16* W2T = (_Float16*)take((size_t)NSF_L * NSF_HID * NSF_HID * 2);
  _Float16* W3T = (_Float16*)take((size_t)NSF_L * 32 * NSF_HID * 2);
  _Float16* h1  = (_Float16*)take((size_t)N * NSF_HID * 2);
  _Float16* h2  = (_Float16*)take((size_t)N * NSF_HID * 2);
  float*    prm = (float*)  take((size_t)N * NSF_P * 4);

  conv_w2_kernel<<<(NSF_L * NSF_HID * NSF_HID + 255) / 256, 256, 0, stream>>>(W2, W2T);
  conv_w3_kernel<<<(NSF_L * 32 * NSF_HID + 255) / 256, 256, 0, stream>>>(W3, W3T);

  // dim 0: broadcast init params
  spline_kernel<<<(N + 255) / 256, 256, 0, stream>>>(x, nullptr, initp, 1, 0, N, out);

  dim3 ggrid(N / 128, NSF_HID / 64);     // 256 x 8 blocks
  for (int i = 1; i < NSF_DIM; i++) {
    const int li = i - 1;
    fc1_kernel<<<(N * NSF_HID) / 256, 256, 0, stream>>>(x, W1, b1, h1, li, i, N);
    gemm512_tanh_kernel<<<ggrid, 256, 0, stream>>>(
        h1, W2T + (size_t)li * NSF_HID * NSF_HID, b2 + li * NSF_HID, h2);
    gemm_w3_kernel<<<N / 128, 256, 0, stream>>>(
        h2, W3T + (size_t)li * 32 * NSF_HID, b3 + li * NSF_P, prm);
    spline_kernel<<<(N + 255) / 256, 256, 0, stream>>>(x, prm, nullptr, 0, i, N, out);
  }
}
